// SemanticSemanticAggregator_79482664780155
// MI455X (gfx1250) — compile-verified, matrix operand order checked by reading, not run on hardware
//
#include <hip/hip_runtime.h>

#define B_  8
#define N_  1024
#define D_  256
#define H_  512
#define M_  256
#define O_  256
#define G4H 2048   // 4*H
#define NWG 16     // workgroups cooperating on the LSTM scan

typedef unsigned short u16;
typedef __attribute__((ext_vector_type(16))) __bf16 v16bf;
typedef __attribute__((ext_vector_type(8)))  float  v8f;
typedef __attribute__((ext_vector_type(8)))  unsigned int v8u;

// round-to-nearest-even fp32 -> bf16
__device__ __forceinline__ u16 f2bf(float f) {
  unsigned u = __builtin_bit_cast(unsigned, f);
  u += 0x7FFFu + ((u >> 16) & 1u);
  return (u16)(u >> 16);
}

// Load one 16x32 bf16 WMMA operand fragment from a row-major [16 x ld] tile.
// CDNA5 16-bit A-layout: lane (r=lane&15, half=lane>>4) holds K-chunks at
// 8*half and 16+8*half. Two b128 loads per lane.
__device__ __forceinline__ v16bf load_frag_rows(const u16* tile, long ld, int kbase, int lane) {
  const int r = lane & 15, hf = (lane >> 4) & 1;
  const u16* p = tile + (long)r * ld + kbase + 8 * hf;
  uint4 a = *(const uint4*)p;
  uint4 b = *(const uint4*)(p + 16);
  v8u u;
  u[0] = a.x; u[1] = a.y; u[2] = a.z; u[3] = a.w;
  u[4] = b.x; u[5] = b.y; u[6] = b.z; u[7] = b.w;
  return __builtin_bit_cast(v16bf, u);
}

#define WMMA_BF16(A, Bf, C) \
  __builtin_amdgcn_wmma_f32_16x16x32_bf16(false, (A), false, (Bf), (short)0, (C), false, false)

// D[M,N](f32) = act( A[M,K](bf16) @ Bm[N,K](bf16)^T + bias ) -- "NT" GEMM.
// One wave computes a 32x32 output block (2x2 register tiles: each fragment
// feeds two independent WMMAs -> 16 FLOP/byte of operand traffic).
// grid = (N/32, M/32, batch). flags: 1 = ReLU, 2 = accumulate into D.
__global__ __launch_bounds__(32) void wmma_gemm_nt(
    const u16* __restrict__ A,  long ldA, long sA,
    const u16* __restrict__ Bm, long ldB, long sB,
    float* __restrict__ D,      long ldD, long sD,
    const float* __restrict__ bias, int K, int flags) {
  const int lane = threadIdx.x;
  const long m0 = (long)blockIdx.y * 32;
  const long n0 = (long)blockIdx.x * 32;
  const u16* At = A + sA * blockIdx.z + m0 * ldA;
  const u16* Bt = Bm + sB * blockIdx.z + n0 * ldB;
  float* Dt = D + sD * blockIdx.z + m0 * ldD + n0;
  const int n = lane & 15, hf = lane >> 4;
  v8f acc[2][2];
  if (flags & 2) {
#pragma unroll
    for (int mi = 0; mi < 2; ++mi)
#pragma unroll
      for (int ni = 0; ni < 2; ++ni)
#pragma unroll
        for (int r = 0; r < 8; ++r)
          acc[mi][ni][r] = Dt[(long)(mi * 16 + r + 8 * hf) * ldD + ni * 16 + n];
  } else {
    float bv0 = bias ? bias[n0 + n] : 0.0f;
    float bv1 = bias ? bias[n0 + 16 + n] : 0.0f;
#pragma unroll
    for (int mi = 0; mi < 2; ++mi)
#pragma unroll
      for (int r = 0; r < 8; ++r) { acc[mi][0][r] = bv0; acc[mi][1][r] = bv1; }
  }
  for (int k = 0; k < K; k += 32) {
    v16bf a0 = load_frag_rows(At, ldA, k, lane);
    v16bf a1 = load_frag_rows(At + 16 * ldA, ldA, k, lane);
    v16bf b0 = load_frag_rows(Bt, ldB, k, lane);
    v16bf b1 = load_frag_rows(Bt + 16 * ldB, ldB, k, lane);
    acc[0][0] = WMMA_BF16(a0, b0, acc[0][0]);
    acc[0][1] = WMMA_BF16(a0, b1, acc[0][1]);
    acc[1][0] = WMMA_BF16(a1, b0, acc[1][0]);
    acc[1][1] = WMMA_BF16(a1, b1, acc[1][1]);
  }
  if (flags & 1) {
#pragma unroll
    for (int mi = 0; mi < 2; ++mi)
#pragma unroll
      for (int ni = 0; ni < 2; ++ni)
#pragma unroll
        for (int r = 0; r < 8; ++r) acc[mi][ni][r] = fmaxf(acc[mi][ni][r], 0.0f);
  }
#pragma unroll
  for (int mi = 0; mi < 2; ++mi)
#pragma unroll
    for (int ni = 0; ni < 2; ++ni)
#pragma unroll
      for (int r = 0; r < 8; ++r)
        Dt[(long)(mi * 16 + r + 8 * hf) * ldD + ni * 16 + n] = acc[mi][ni][r];
}

// ---------------------------------------------------------------------------
// Cooperative LSTM scan across NWG=16 workgroups (one per WGP).
// WG w owns h-columns [w*32, w*32+32); its gate-reordered W_hh slice (all 4
// gates of those columns: 4 groups x 32 rows x 512 = 128 KB bf16) is staged
// into LDS by the Tensor Data Mover as a 3D tile (tile_dim0=512 elements,
// tile_dim1=32 rows/gate, tile_dim2=4 gates; dim1_stride = 512*512 hops
// between gate groups) and stays resident for the whole scan.
// Per step: 8 waves x 16 WMMAs (A = h from LDS, B = W_hh slice from LDS),
// WG-local cell update, then an 8 KB h broadcast through L2 with a
// double-buffered device-scope arrive/spin barrier (s_sleep in the spin).
// LDS: gates 8x128 f32 (4 KB) | Wslice 128x512 bf16 (128 KB) | h 16x512 bf16 (16 KB)
// ---------------------------------------------------------------------------
__global__ __launch_bounds__(256) void lstm_scan_mw(
    const float* __restrict__ xproj,  // [B, N, 4H] = x@W_ih^T + (b_ih+b_hh)
    const u16*  __restrict__ Whh,     // [4H, H] bf16
    float* __restrict__ h_out,        // [B, N, H]
    u16*  __restrict__ hbf_out,       // [B, N, H]
    u16*  __restrict__ hT_out,        // [B, H, N]
    u16*  __restrict__ hg,            // [2][8][512] bf16 broadcast (double buf)
    unsigned* __restrict__ bar) {     // global arrival counter (zeroed by host)
  extern __shared__ char smem[];
  float* gatesL = (float*)smem;                      // [8][128]
  u16*   Wslice = (u16*)(smem + 4096);               // [128][512]
  u16*   hbuf   = (u16*)(smem + 4096 + 128 * H_ * 2);// [16][512]
  const int tid = threadIdx.x;
  const int wg = blockIdx.x;           // 0..15
  const int hc0 = wg * 32;             // owned h-column base
  const int wv = tid >> 5, lane = tid & 31;
  const int n16 = lane & 15, hf = lane >> 4;

  // ---- Stage gate-reordered W_hh slice: LDS row (g*32+j) <- row g*512+hc0+j
#if __has_builtin(__builtin_amdgcn_tensor_load_to_lds)
  if (tid < 32) {  // one wave issues the TDM descriptor (D# per ISA ch.8)
    typedef unsigned int u32x4 __attribute__((ext_vector_type(4)));
    typedef int          i32x4 __attribute__((ext_vector_type(4)));
    typedef int          i32x8 __attribute__((ext_vector_type(8)));
    const unsigned long long ga = (unsigned long long)(const void*)(Whh + (long)hc0 * H_);
    const unsigned ldsoff = (unsigned)(unsigned long long)(void*)Wslice; // LDS aperture: low 32 bits
    u32x4 g0;
    g0[0] = 1u;                                                  // count=1, user D#
    g0[1] = ldsoff;                                              // lds_addr
    g0[2] = (unsigned)(ga & 0xFFFFFFFFull);                      // global_addr[31:0]
    g0[3] = (unsigned)((ga >> 32) & 0x01FFFFFFull) | (2u << 30); // addr[56:32] | type=2
    i32x8 g1;
    g1[0] = (int)(1u << 16);           // workgroup_mask=0 | data_size=1 (2 B)
    g1[1] = (int)(512u << 16);         // tensor_dim0 = 512 (row length, elems)
    g1[2] = (int)(512u << 16);         // tensor_dim1 = 512 (rows per gate group)
    g1[3] = (int)(512u << 16);         // tile_dim0 = 512
    g1[4] = (int)(32u | (4u << 16));   // tile_dim1 = 32, tile_dim2 = 4 gates
    g1[5] = 512;                       // tensor_dim0_stride = 512 elems
    g1[6] = 0;                         // stride hi bits / dim1_stride[15:0]
    g1[7] = 4;                         // tensor_dim1_stride = 4<<16 = 262144 (gate hop)
    i32x4 g2; g2[0] = 4; g2[1] = 0; g2[2] = 0; g2[3] = 0;  // tensor_dim2 = 4
    i32x4 g3; g3[0] = 0; g3[1] = 0; g3[2] = 0; g3[3] = 0;
#if __clang_major__ >= 23
    i32x8 g4 = {0, 0, 0, 0, 0, 0, 0, 0};
    __builtin_amdgcn_tensor_load_to_lds(g0, g1, g2, g3, g4, 0);
#else
    __builtin_amdgcn_tensor_load_to_lds(g0, g1, g2, g3, 0);
#endif
    __builtin_amdgcn_s_wait_tensorcnt(0);
  }
#else
  {
    uint4* dW = (uint4*)Wslice;        // 128 rows x 64 uint4
    for (int idx = tid; idx < 128 * 64; idx += 256) {
      const int rr = idx >> 6, cq = idx & 63;
      const uint4* srow =
          (const uint4*)(Whh + ((long)(rr >> 5) * H_ + hc0 + (rr & 31)) * H_);
      dW[idx] = srow[cq];
    }
  }
#endif
  for (int i = tid; i < 16 * H_; i += 256) hbuf[i] = 0;   // h[-1] = 0 (+pad rows)
  float creg = 0.0f;                                      // c for (b=tid>>5, hc0+(tid&31))
  __syncthreads();

  for (int s = 0; s < N_; ++s) {
    // ---- gates(owned 128 cols) = xproj[:, s, reordered] + h @ Wslice^T ----
    {
      const int lt0 = wv * 16;          // this wave's 16 local gate columns
      v8f acc;
#pragma unroll
      for (int r = 0; r < 8; ++r) {
        const int m = r + 8 * hf;
        const int lc = lt0 + n16;
        const int gcol = (lc >> 5) * H_ + hc0 + (lc & 31);
        acc[r] = (m < 8) ? xproj[((long)m * N_ + s) * G4H + gcol] : 0.0f;
      }
      const u16* Bt = Wslice + (long)lt0 * H_;
      for (int k = 0; k < H_; k += 32) {
        v16bf af = load_frag_rows(hbuf, H_, k, lane);
        v16bf bf = load_frag_rows(Bt, H_, k, lane);
        acc = WMMA_BF16(af, bf, acc);
      }
      if (hf == 0) {
#pragma unroll
        for (int r = 0; r < 8; ++r) gatesL[r * 128 + lt0 + n16] = acc[r];
      }
    }
    __syncthreads();
    // ---- cell update for the 256 owned (b, hc) pairs (one per thread) ----
    {
      const int b = tid >> 5, j = tid & 31;
      const int hc = hc0 + j;
      const float gi = gatesL[b * 128 + j];
      const float gf = gatesL[b * 128 + 32 + j];
      const float gg = gatesL[b * 128 + 64 + j];
      const float go = gatesL[b * 128 + 96 + j];
      const float fi = 1.0f / (1.0f + __expf(-gi));
      const float ff = 1.0f / (1.0f + __expf(-gf));
      const float fo = 1.0f / (1.0f + __expf(-go));
      creg = ff * creg + fi * tanhf(gg);
      const float hv = fo * tanhf(creg);
      const long base = ((long)b * N_ + s) * H_ + hc;
      h_out[base] = hv;
      const u16 hb = f2bf(hv);
      hbf_out[base] = hb;
      hT_out[((long)b * H_ + hc) * N_ + s] = hb;
      hg[(size_t)(s & 1) * (8 * H_) + b * H_ + hc] = hb;   // broadcast slice
    }
    __threadfence();
    __syncthreads();
    // ---- one global barrier per step (arrive + spin on tid 0) ----
    if (tid == 0) {
      atomicAdd(bar, 1u);
      const unsigned target = (unsigned)NWG * (unsigned)(s + 1);
      while (__hip_atomic_load(bar, __ATOMIC_ACQUIRE, __HIP_MEMORY_SCOPE_AGENT) < target)
        __builtin_amdgcn_s_sleep(1);
    }
    __syncthreads();
    __threadfence();
    // ---- refresh full h state from the step-s broadcast buffer (b128) ----
    {
      const uint4* sp = (const uint4*)(hg + (size_t)(s & 1) * (8 * H_));
      uint4* dp = (uint4*)hbuf;
      for (int i = tid; i < (8 * H_) / 8; i += 256) dp[i] = sp[i];
    }
    __syncthreads();
  }
}

// Row-wise masked softmax over scores[b,i,:], mask = (adj>0 && j!=i).
__global__ __launch_bounds__(256) void masked_softmax(
    const float* __restrict__ scores, const int* __restrict__ adj,
    u16* __restrict__ w_bf, int* __restrict__ has_nb) {
  const int row = blockIdx.x;          // b*N + i
  const int i = row & (N_ - 1);
  const float* srow = scores + (long)row * N_;
  const int* arow = adj + (long)row * N_;
  const int t = threadIdx.x;
  __shared__ float red[256];
  float vals[4]; int msk[4];
  float lmax = -3.0e38f;
#pragma unroll
  for (int q = 0; q < 4; ++q) {
    const int j = t + q * 256;
    msk[q] = (arow[j] > 0) && (j != i);
    vals[q] = srow[j];
    if (msk[q]) lmax = fmaxf(lmax, vals[q]);
  }
  red[t] = lmax; __syncthreads();
  for (int s2 = 128; s2 > 0; s2 >>= 1) {
    if (t < s2) red[t] = fmaxf(red[t], red[t + s2]);
    __syncthreads();
  }
  const float rmax = red[0]; __syncthreads();
  float e[4]; float lsum = 0.f;
#pragma unroll
  for (int q = 0; q < 4; ++q) { e[q] = msk[q] ? __expf(vals[q] - rmax) : 0.0f; lsum += e[q]; }
  red[t] = lsum; __syncthreads();
  for (int s2 = 128; s2 > 0; s2 >>= 1) {
    if (t < s2) red[t] += red[t + s2];
    __syncthreads();
  }
  const float rsum = red[0];
  const int any = rmax > -2.9e38f;
  const float inv = any ? (1.0f / rsum) : 0.0f;
#pragma unroll
  for (int q = 0; q < 4; ++q) w_bf[(long)row * N_ + t + q * 256] = f2bf(e[q] * inv);
  if (t == 0) has_nb[row] = any;
}

__global__ void cvt_f32_bf16(const float* __restrict__ src, u16* __restrict__ dst, long n) {
  const long i = (long)blockIdx.x * blockDim.x + threadIdx.x;
  if (i < n) dst[i] = f2bf(src[i]);
}

__global__ void add_bias(const float* __restrict__ a, const float* __restrict__ b,
                         float* __restrict__ c, int n) {
  const int i = blockIdx.x * blockDim.x + threadIdx.x;
  if (i < n) c[i] = a[i] + b[i];
}

// agg rows with no neighbors -> 0; convert to bf16 for the final GEMM.
__global__ __launch_bounds__(256) void mask_cvt(const float* __restrict__ src,
                                                const int* __restrict__ flags,
                                                u16* __restrict__ dst) {
  const int row = blockIdx.x;
  const int t = threadIdx.x;
  const float v = flags[row] ? src[(long)row * O_ + t] : 0.0f;
  dst[(long)row * O_ + t] = f2bf(v);
}

extern "C" void kernel_launch(void* const* d_in, const int* in_sizes, int n_in,
                              void* d_out, int out_size, void* d_ws, size_t ws_size,
                              hipStream_t stream) {
  (void)in_sizes; (void)n_in; (void)out_size; (void)ws_size;
  const float* feats = (const float*)d_in[0];
  const int*   adj   = (const int*)d_in[1];
  const float* W_ih  = (const float*)d_in[2];
  const float* W_hh  = (const float*)d_in[3];
  const float* b_ih  = (const float*)d_in[4];
  const float* b_hh  = (const float*)d_in[5];
  const float* gx_W1 = (const float*)d_in[6];  const float* gx_b1 = (const float*)d_in[7];
  const float* gx_W2 = (const float*)d_in[8];  const float* gx_b2 = (const float*)d_in[9];
  const float* gz_W1 = (const float*)d_in[10]; const float* gz_b1 = (const float*)d_in[11];
  const float* gz_W2 = (const float*)d_in[12]; const float* gz_b2 = (const float*)d_in[13];
  const float* gv_W1 = (const float*)d_in[14]; const float* gv_b1 = (const float*)d_in[15];
  const float* gv_W2 = (const float*)d_in[16]; const float* gv_b2 = (const float*)d_in[17];
  const float* gn_W1 = (const float*)d_in[18]; const float* gn_b1 = (const float*)d_in[19];
  const float* gn_W2 = (const float*)d_in[20]; const float* gn_b2 = (const float*)d_in[21];
  const float* out_W = (const float*)d_in[22]; const float* out_b = (const float*)d_in[23];
  float* out = (float*)d_out;

  char* ws = (char*)d_ws;
  size_t off = 0;
  auto alloc = [&](size_t bytes) { size_t o = off; off += (bytes + 255) & ~(size_t)255; return o; };
  const long BN = (long)B_ * N_;  // 8192

  u16* feats_bf = (u16*)(ws + alloc((size_t)BN * D_ * 2));
  u16* Wih_bf   = (u16*)(ws + alloc((size_t)G4H * D_ * 2));
  u16* Whh_bf   = (u16*)(ws + alloc((size_t)G4H * H_ * 2));
  u16* gxW1_bf  = (u16*)(ws + alloc((size_t)M_ * H_ * 2));
  u16* gxW2_bf  = (u16*)(ws + alloc((size_t)M_ * M_ * 2));
  u16* gvW1_bf  = (u16*)(ws + alloc((size_t)M_ * H_ * 2));
  u16* gvW2_bf  = (u16*)(ws + alloc((size_t)M_ * M_ * 2));
  u16* gzW1_bf  = (u16*)(ws + alloc((size_t)M_ * H_ * 2));
  u16* gzW2_bf  = (u16*)(ws + alloc((size_t)M_ * M_ * 2));
  u16* gnW1_bf  = (u16*)(ws + alloc((size_t)M_ * M_ * 2));
  u16* gnW2_bf  = (u16*)(ws + alloc((size_t)O_ * M_ * 2));
  u16* outW_bf  = (u16*)(ws + alloc((size_t)O_ * (H_ + O_) * 2));
  float* bsum   = (float*)(ws + alloc((size_t)G4H * 4));
  int* has_nb   = (int*)(ws + alloc((size_t)BN * 4));
  unsigned* bar = (unsigned*)(ws + alloc(256));
  u16* hg       = (u16*)(ws + alloc((size_t)2 * 8 * H_ * 2));
  float* h      = (float*)(ws + alloc((size_t)BN * H_ * 4));
  u16* h_bf     = (u16*)(ws + alloc((size_t)BN * H_ * 2));
  u16* hT_bf    = (u16*)(ws + alloc((size_t)BN * H_ * 2));
  // 64 MB region: xproj first, then reused for scores (32 MB) + w_bf (16 MB)
  size_t xoff   = alloc((size_t)BN * G4H * 4);
  float* xproj  = (float*)(ws + xoff);
  float* scores = (float*)(ws + xoff);
  u16* w_bf     = (u16*)(ws + xoff + (size_t)B_ * N_ * N_ * 4);
  float* t1     = (float*)(ws + alloc((size_t)BN * M_ * 4));
  u16* t1_bf    = (u16*)(ws + alloc((size_t)BN * M_ * 2));
  float* ax     = (float*)(ws + alloc((size_t)BN * M_ * 4));
  u16* ax_bf    = (u16*)(ws + alloc((size_t)BN * M_ * 2));
  float* av     = (float*)(ws + alloc((size_t)BN * M_ * 4));
  u16* av_bf    = (u16*)(ws + alloc((size_t)BN * M_ * 2));
  float* sagg   = (float*)(ws + alloc((size_t)BN * H_ * 4));
  u16* sagg_bf  = (u16*)(ws + alloc((size_t)BN * H_ * 2));
  float* agg1   = (float*)(ws + alloc((size_t)BN * M_ * 4));
  u16* agg1_bf  = (u16*)(ws + alloc((size_t)BN * M_ * 2));
  float* agg    = (float*)(ws + alloc((size_t)BN * O_ * 4));
  u16* agg_bf   = (u16*)(ws + alloc((size_t)BN * O_ * 2));

  const dim3 b256(256), b32(32);
  auto cvt = [&](const float* s, u16* d, long n) {
    cvt_f32_bf16<<<dim3((unsigned)((n + 255) / 256)), b256, 0, stream>>>(s, d, n);
  };
  auto gemm = [&](const u16* A, long ldA, long sA, const u16* Bm, long ldB, long sB,
                  float* D, long ldD, long sD, const float* bias,
                  int Mrows, int Ncols, int K, int batch, int flags) {
    wmma_gemm_nt<<<dim3(Ncols / 32, Mrows / 32, batch), b32, 0, stream>>>(
        A, ldA, sA, Bm, ldB, sB, D, ldD, sD, bias, K, flags);
  };

  // ---- bf16 conversions of activation-invariant data ----
  cvt(feats, feats_bf, BN * D_);
  cvt(W_ih,  Wih_bf,  (long)G4H * D_);
  cvt(W_hh,  Whh_bf,  (long)G4H * H_);
  cvt(gx_W1, gxW1_bf, (long)M_ * H_);
  cvt(gx_W2, gxW2_bf, (long)M_ * M_);
  cvt(gv_W1, gvW1_bf, (long)M_ * H_);
  cvt(gv_W2, gvW2_bf, (long)M_ * M_);
  cvt(gz_W1, gzW1_bf, (long)M_ * H_);
  cvt(gz_W2, gzW2_bf, (long)M_ * M_);
  cvt(gn_W1, gnW1_bf, (long)M_ * M_);
  cvt(gn_W2, gnW2_bf, (long)O_ * M_);
  cvt(out_W, outW_bf, (long)O_ * (H_ + O_));
  add_bias<<<dim3((G4H + 255) / 256), b256, 0, stream>>>(b_ih, b_hh, bsum, G4H);

  // ---- x_proj = feats @ W_ih^T + (b_ih+b_hh)   [8192 x 2048], K=256 ----
  gemm(feats_bf, D_, 0, Wih_bf, D_, 0, xproj, G4H, 0, bsum, (int)BN, G4H, D_, 1, 0);

  // ---- cooperative LSTM scan: 16 WGs, TDM-staged LDS-resident W_hh ----
  hipMemsetAsync(bar, 0, sizeof(unsigned), stream);
  hipFuncSetAttribute((const void*)lstm_scan_mw,
                      hipFuncAttributeMaxDynamicSharedMemorySize,
                      4096 + 128 * H_ * 2 + 16 * H_ * 2);
  lstm_scan_mw<<<dim3(NWG), b256, 4096 + 128 * H_ * 2 + 16 * H_ * 2, stream>>>(
      xproj, Whh_bf, h, h_bf, hT_bf, hg, bar);

  // ---- ax = mlp_gx(h), av = mlp_gv(h) ----
  gemm(h_bf, H_, 0, gxW1_bf, H_, 0, t1, M_, 0, gx_b1, (int)BN, M_, H_, 1, 1);
  cvt(t1, t1_bf, BN * M_);
  gemm(t1_bf, M_, 0, gxW2_bf, M_, 0, ax, M_, 0, gx_b2, (int)BN, M_, M_, 1, 0);
  cvt(ax, ax_bf, BN * M_);
  gemm(h_bf, H_, 0, gvW1_bf, H_, 0, t1, M_, 0, gv_b1, (int)BN, M_, H_, 1, 1);
  cvt(t1, t1_bf, BN * M_);
  gemm(t1_bf, M_, 0, gvW2_bf, M_, 0, av, M_, 0, gv_b2, (int)BN, M_, M_, 1, 0);
  cvt(av, av_bf, BN * M_);

  // ---- scores[b] = ax[b] @ av[b]^T   (batched NT, K=256) ----
  gemm(ax_bf, M_, (long)N_ * M_, av_bf, M_, (long)N_ * M_,
       scores, N_, (long)N_ * N_, nullptr, N_, N_, M_, B_, 0);

  // ---- masked softmax over neighbors -> w (bf16) + has_nb ----
  masked_softmax<<<dim3((unsigned)BN), b256, 0, stream>>>(scores, adj, w_bf, has_nb);

  // ---- sagg[b] = w[b] @ h[b]  via h^T (NT, K=1024) ----
  gemm(w_bf, N_, (long)N_ * N_, hT_bf, N_, (long)H_ * N_,
       sagg, H_, (long)N_ * H_, nullptr, N_, H_, N_, B_, 0);
  cvt(sagg, sagg_bf, BN * H_);

  // ---- agg = mlp_gn(mlp_gz(sagg)) ----
  gemm(sagg_bf, H_, 0, gzW1_bf, H_, 0, t1, M_, 0, gz_b1, (int)BN, M_, H_, 1, 1);
  cvt(t1, t1_bf, BN * M_);
  gemm(t1_bf, M_, 0, gzW2_bf, M_, 0, agg1, M_, 0, gz_b2, (int)BN, M_, M_, 1, 0);
  cvt(agg1, agg1_bf, BN * M_);
  gemm(agg1_bf, M_, 0, gnW1_bf, M_, 0, t1, M_, 0, gn_b1, (int)BN, M_, M_, 1, 1);
  cvt(t1, t1_bf, BN * M_);
  gemm(t1_bf, M_, 0, gnW2_bf, M_, 0, agg, O_, 0, gn_b2, (int)BN, O_, M_, 1, 0);
  mask_cvt<<<dim3((unsigned)BN), b256, 0, stream>>>(agg, has_nb, agg_bf);

  // ---- out = [h, agg] @ out_W^T + out_b  (split-K over the concat) ----
  gemm(h_bf, H_, 0, outW_bf, H_ + O_, 0, out, O_, 0, out_b, (int)BN, O_, H_, 1, 0);
  gemm(agg_bf, O_, 0, outW_bf + H_, H_ + O_, 0, out, O_, 0, nullptr, (int)BN, O_, O_, 1, 2);
}
